// ShidoTreeLSTMLayer_13683765805744
// MI455X (gfx1250) — matrix operand order repaired and checked by reading, not compiled, not run on hardware
//
#include <hip/hip_runtime.h>
#include <hip/hip_bf16.h>
#include <math.h>

// ---------------------------------------------------------------------------
// ShidoTreeLSTMLayer for MI455X (gfx1250), wave32 + WMMA f32_16x16x32_f16.
// All GEMM B-operands are weights, pre-transposed to N-major once per launch
// so both A and B tiles stage as contiguous 16B chunks (async-LDS eligible).
// Workspace requirement: ~230 MB.
// ---------------------------------------------------------------------------

#define L_LEVELS 6
#define N_NODES  4096
#define MAX_CH   8
#define DIM      256      // DIM_IN == DIM_OUT == 256
#define GATED    1024     // 4*DIM
#define LDSP     40       // padded LDS row stride (halfs): 80B, 16B aligned

#ifndef __has_builtin
#define __has_builtin(x) 0
#endif

#if __has_builtin(__builtin_amdgcn_global_load_async_to_lds_b128)
#define HAVE_ASYNC_LDS 1
#else
#define HAVE_ASYNC_LDS 0
#endif

typedef __attribute__((ext_vector_type(16))) _Float16 v16h;
typedef __attribute__((ext_vector_type(8)))  _Float16 v8h;
typedef __attribute__((ext_vector_type(8)))  float    v8f;
typedef __attribute__((ext_vector_type(4)))  int      v4i;

__device__ __forceinline__ float sigmoidf_(float x) {
    return 1.0f / (1.0f + __expf(-x));
}

#if HAVE_ASYNC_LDS
// b128 async copy: params are payload-typed (v4i*) pointers per the builtin.
__device__ __forceinline__ void async_cp16(const _Float16* g, _Float16* l) {
    __builtin_amdgcn_global_load_async_to_lds_b128(
        (__attribute__((address_space(1))) v4i*)g,
        (__attribute__((address_space(3))) v4i*)l, 0, 0);
}
__device__ __forceinline__ void wait_async_le4() {
#if __has_builtin(__builtin_amdgcn_s_wait_asynccnt)
    __builtin_amdgcn_s_wait_asynccnt(4);
#else
    asm volatile("s_wait_asynccnt 0x4" ::: "memory");
#endif
}
__device__ __forceinline__ void wait_async_0() {
#if __has_builtin(__builtin_amdgcn_s_wait_asynccnt)
    __builtin_amdgcn_s_wait_asynccnt(0);
#else
    asm volatile("s_wait_asynccnt 0x0" ::: "memory");
#endif
}
#endif

// ---------------------------------------------------------------------------
// One-time staging kernels
// ---------------------------------------------------------------------------
__global__ void cvt_f32_f16(const float* __restrict__ src,
                            _Float16* __restrict__ dst, int n) {
    int i = blockIdx.x * blockDim.x + threadIdx.x;
    if (i < n) dst[i] = (_Float16)src[i];
}

// src: f32 [Ks][Ns] row-major  ->  dst: f16 [Ns][ldd] with dst[n*ldd+kofs+k]
__global__ void cvt_transpose_f16(const float* __restrict__ src,
                                  _Float16* __restrict__ dst,
                                  int Ks, int Ns, int ldd, int kofs) {
    int i = blockIdx.x * blockDim.x + threadIdx.x;
    if (i >= Ks * Ns) return;
    int k = i / Ns, n = i - k * Ns;              // coalesced source read
    dst[(size_t)n * ldd + kofs + k] = (_Float16)src[i];
}

__global__ void zero_row0(float* __restrict__ htab, float* __restrict__ ctab) {
    int d = threadIdx.x;            // 256 threads
    htab[d] = 0.0f;
    ctab[d] = 0.0f;
}

// ---------------------------------------------------------------------------
// WMMA GEMM:  C[M,N] (f32) = concat_K(A1[0:K1], A2[K1:K]) (f16) @ B (f16)
//             (+ bias[N] if non-null)
// B is pre-transposed: BT[n][k], leading dim ldbt.
// Block: 128 threads (4 waves). Block tile 64x64, wave tile 16x64,
// K-tiles of 32 matching v_wmma_f32_16x16x32_f16. Async double-buffered LDS.
// ---------------------------------------------------------------------------
#if HAVE_ASYNC_LDS
#define NBUF 2
#else
#define NBUF 1
#endif

__global__ __launch_bounds__(128)
void gemm_f16_wmma(const _Float16* __restrict__ A1, int lda1, int K1,
                   const _Float16* __restrict__ A2, int lda2,
                   const _Float16* __restrict__ BT, int ldbt,
                   const float* __restrict__ bias,
                   float* __restrict__ C, int ldc,
                   int M, int N, int K) {
    __shared__ _Float16 As[NBUF][64][LDSP];
    __shared__ _Float16 Bs[NBUF][64][LDSP];

    const int tid   = threadIdx.x;
    const int lane  = tid & 31;
    const int wv    = tid >> 5;       // 0..3
    const int half  = lane >> 4;      // 0/1
    const int l16   = lane & 15;

    const int tileM = blockIdx.y * 64;
    const int tileN = blockIdx.x * 64;

    // Staging geometry: tile = 64 rows x 32 halfs = 256 x 16B chunks,
    // 2 chunks per thread (rows r0 and r0+32), all pointer math hoisted.
    const int r0 = tid >> 2;                 // 0..31
    const int c0 = (tid & 3) * 8;            // half offset 0/8/16/24
    const _Float16* a1p0 = A1 + (size_t)(tileM + r0)      * lda1 + c0;
    const _Float16* a1p1 = A1 + (size_t)(tileM + r0 + 32) * lda1 + c0;
    const _Float16* a2p0 = A2 + (size_t)(tileM + r0)      * lda2 + c0;
    const _Float16* a2p1 = A2 + (size_t)(tileM + r0 + 32) * lda2 + c0;
    const _Float16* bp0  = BT + (size_t)(tileN + r0)      * ldbt + c0;
    const _Float16* bp1  = BT + (size_t)(tileN + r0 + 32) * ldbt + c0;

    v8f acc[4];
    #pragma unroll
    for (int j = 0; j < 4; ++j) acc[j] = (v8f)(0.0f);

    const int arow = wv * 16 + l16;

#if HAVE_ASYNC_LDS
    // ---- async double-buffered pipeline --------------------------------
    auto stage = [&](int buf, int kbase) {
        const _Float16 *ga0, *ga1;
        if (kbase < K1) { ga0 = a1p0 + kbase;      ga1 = a1p1 + kbase; }
        else { int o = kbase - K1; ga0 = a2p0 + o; ga1 = a2p1 + o; }
        async_cp16(ga0,        &As[buf][r0][c0]);
        async_cp16(ga1,        &As[buf][r0 + 32][c0]);
        async_cp16(bp0 + kbase, &Bs[buf][r0][c0]);
        async_cp16(bp1 + kbase, &Bs[buf][r0 + 32][c0]);
    };

    stage(0, 0);
    int buf = 0;
    for (int kbase = 0; kbase < K; kbase += 32) {
        const int nk = kbase + 32;
        const bool more = nk < K;
        if (more) stage(buf ^ 1, nk);            // next tile in flight
        if (more) wait_async_le4();              // oldest 4 (this tile) done
        else      wait_async_0();
        __syncthreads();

        v8h alo = *(const v8h*)&As[buf][arow][8 * half];
        v8h ahi = *(const v8h*)&As[buf][arow][16 + 8 * half];
        v16h af = __builtin_shufflevector(alo, ahi,
            0, 1, 2, 3, 4, 5, 6, 7, 8, 9, 10, 11, 12, 13, 14, 15);
        #pragma unroll
        for (int j = 0; j < 4; ++j) {
            int bcol = j * 16 + l16;
            v8h b0 = *(const v8h*)&Bs[buf][bcol][8 * half];
            v8h b1 = *(const v8h*)&Bs[buf][bcol][16 + 8 * half];
            v16h bf = __builtin_shufflevector(b0, b1,
                0, 1, 2, 3, 4, 5, 6, 7, 8, 9, 10, 11, 12, 13, 14, 15);
            acc[j] = __builtin_amdgcn_wmma_f32_16x16x32_f16(
                false, af, false, bf, (short)0, acc[j], false, false);
        }
        if (more) __syncthreads();               // before restaging this buf
        buf ^= 1;
    }
#else
    // ---- sync fallback: issue all loads, barrier, store, barrier -------
    for (int kbase = 0; kbase < K; kbase += 32) {
        const _Float16 *ga0, *ga1;
        if (kbase < K1) { ga0 = a1p0 + kbase;      ga1 = a1p1 + kbase; }
        else { int o = kbase - K1; ga0 = a2p0 + o; ga1 = a2p1 + o; }
        // loads issue back-to-back and overlap the barrier wait
        v8h va0 = *(const v8h*)ga0;
        v8h va1 = *(const v8h*)ga1;
        v8h vb0 = *(const v8h*)(bp0 + kbase);
        v8h vb1 = *(const v8h*)(bp1 + kbase);
        int nk = kbase + 32;
        if (nk < K) {                            // prefetch next tile
            const _Float16* pa = (nk < K1) ? a1p0 + nk : a2p0 + (nk - K1);
            __builtin_prefetch(pa, 0, 1);
            __builtin_prefetch(bp0 + nk, 0, 1);
        }
        __syncthreads();                         // prev compute done
        *(v8h*)&As[0][r0][c0]      = va0;
        *(v8h*)&As[0][r0 + 32][c0] = va1;
        *(v8h*)&Bs[0][r0][c0]      = vb0;
        *(v8h*)&Bs[0][r0 + 32][c0] = vb1;
        __syncthreads();

        v8h alo = *(const v8h*)&As[0][arow][8 * half];
        v8h ahi = *(const v8h*)&As[0][arow][16 + 8 * half];
        v16h af = __builtin_shufflevector(alo, ahi,
            0, 1, 2, 3, 4, 5, 6, 7, 8, 9, 10, 11, 12, 13, 14, 15);
        #pragma unroll
        for (int j = 0; j < 4; ++j) {
            int bcol = j * 16 + l16;
            v8h b0 = *(const v8h*)&Bs[0][bcol][8 * half];
            v8h b1 = *(const v8h*)&Bs[0][bcol][16 + 8 * half];
            v16h bf = __builtin_shufflevector(b0, b1,
                0, 1, 2, 3, 4, 5, 6, 7, 8, 9, 10, 11, 12, 13, 14, 15);
            acc[j] = __builtin_amdgcn_wmma_f32_16x16x32_f16(
                false, af, false, bf, (short)0, acc[j], false, false);
        }
    }
#endif

    // ---- epilogue: C/D layout -> row = wv*16 + 8*half + v, col = l16 ----
    #pragma unroll
    for (int j = 0; j < 4; ++j) {
        int col = tileN + j * 16 + l16;
        float bv = bias ? bias[col] : 0.0f;
        #pragma unroll
        for (int v = 0; v < 8; ++v) {
            int row = tileM + wv * 16 + half * 8 + v;
            C[(size_t)row * ldc + col] = acc[j][v] + bv;
        }
    }
}

// ---------------------------------------------------------------------------
// Gather children states + build reversed sequence + lengths.
// ---------------------------------------------------------------------------
__global__ __launch_bounds__(256)
void build_seq(const int* __restrict__ idx,
               const float* __restrict__ htab, const float* __restrict__ ctab,
               _Float16* __restrict__ hs, _Float16* __restrict__ hrev,
               float* __restrict__ cs, int* __restrict__ length) {
    int b = blockIdx.x;
    int d = threadIdx.x;
    int iv[MAX_CH];
    int len = 0;
    #pragma unroll
    for (int j = 0; j < MAX_CH; ++j) {
        iv[j] = idx[b * MAX_CH + j];
        len += (iv[j] != -1) ? 1 : 0;
    }
    #pragma unroll
    for (int t = 0; t < MAX_CH; ++t) {
        int g = iv[t] < 0 ? 0 : iv[t];
        size_t o = ((size_t)b * MAX_CH + t) * DIM + d;
        hs[o] = (_Float16)htab[(size_t)g * DIM + d];
        cs[o] = ctab[(size_t)g * DIM + d];
        int tr = (t < len) ? (len - 1 - t) : t;    // tf.reverse_sequence
        int gr = iv[tr] < 0 ? 0 : iv[tr];
        hrev[o] = (_Float16)htab[(size_t)gr * DIM + d];
    }
    if (d == 0) length[b] = len;
}

__global__ __launch_bounds__(256)
void init_states(float* __restrict__ h, float* __restrict__ c,
                 _Float16* __restrict__ h16,
                 const float* __restrict__ ih, const float* __restrict__ ic) {
    int b = blockIdx.x, d = threadIdx.x;
    float hv = ih[d], cv = ic[d];
    h[(size_t)b * DIM + d]   = hv;
    c[(size_t)b * DIM + d]   = cv;
    h16[(size_t)b * DIM + d] = (_Float16)hv;
}

// LSTM cell pointwise: gate order i,f,g,o. Writes h into concat y [N*8,512].
__global__ __launch_bounds__(256)
void lstm_cell(const float* __restrict__ Z,
               float* __restrict__ hst, float* __restrict__ cst,
               _Float16* __restrict__ hst16,
               _Float16* __restrict__ y, int ycoloff, int t) {
    int b = blockIdx.x, d = threadIdx.x;
    size_t zb = (size_t)b * GATED;
    float zi = Z[zb + d];
    float zf = Z[zb + DIM + d];
    float zg = Z[zb + 2 * DIM + d];
    float zo = Z[zb + 3 * DIM + d];
    size_t s = (size_t)b * DIM + d;
    float c = sigmoidf_(zf) * cst[s] + sigmoidf_(zi) * tanhf(zg);
    float h = sigmoidf_(zo) * tanhf(c);
    cst[s] = c;
    hst[s] = h;
    hst16[s] = (_Float16)h;
    y[((size_t)b * MAX_CH + t) * (2 * DIM) + ycoloff + d] = (_Float16)h;
}

__global__ __launch_bounds__(256)
void gather_last(const _Float16* __restrict__ y, const int* __restrict__ length,
                 _Float16* __restrict__ ylast) {
    int b = blockIdx.x;
    int j = blockIdx.y * 256 + threadIdx.x;       // 0..511
    int last = length[b] - 1;
    ylast[(size_t)b * (2 * DIM) + j] =
        y[((size_t)b * MAX_CH + last) * (2 * DIM) + j];
}

// Final gate combine. Wx split order: f,i,u,o.
__global__ __launch_bounds__(256)
void combine(const float* __restrict__ Wx,
             const float* __restrict__ fc0,   // [N*8, 256] seq branch
             const float* __restrict__ fc1,   // [N, 256] last-step branches
             const float* __restrict__ fc2,
             const float* __restrict__ fc3,
             const float* __restrict__ cs,    // gathered child c, [N*8,256]
             const int* __restrict__ length,
             float* __restrict__ htab, float* __restrict__ ctab,
             float* __restrict__ out_h, float* __restrict__ out_c) {
    int b = blockIdx.x, d = threadIdx.x;
    int len = length[b];
    size_t wb = (size_t)b * GATED;
    float Wf = Wx[wb + d];
    float Wi = Wx[wb + DIM + d];
    float Wu = Wx[wb + 2 * DIM + d];
    float Wo = Wx[wb + 3 * DIM + d];

    float bf = 0.0f;
    #pragma unroll
    for (int t = 0; t < MAX_CH; ++t) {
        size_t o = ((size_t)b * MAX_CH + t) * DIM + d;
        float s = sigmoidf_(Wf + fc0[o]) * cs[o];
        bf += (t < len) ? s : 0.0f;
    }
    size_t lb = (size_t)b * DIM + d;
    float bi = sigmoidf_(fc1[lb] + Wi);
    float bu = tanhf(fc2[lb] + Wu);
    float bo = sigmoidf_(fc3[lb] + Wo);
    float nc = bi * bu + bf;
    float nh = bo * tanhf(nc);

    out_h[lb] = nh;
    out_c[lb] = nc;
    htab[(size_t)(b + 1) * DIM + d] = nh;   // row 0 stays zero
    ctab[(size_t)(b + 1) * DIM + d] = nc;
}

// ---------------------------------------------------------------------------
// Host orchestration
// ---------------------------------------------------------------------------
extern "C" void kernel_launch(void* const* d_in, const int* in_sizes, int n_in,
                              void* d_out, int out_size, void* d_ws, size_t ws_size,
                              hipStream_t stream) {
    (void)in_sizes; (void)n_in; (void)out_size; (void)ws_size;

    const float* tensor    = (const float*)d_in[0];   // [6,4096,256]
    const int*   indices   = (const int*)  d_in[1];   // [6,4096,8]
    const float* W_kernel  = (const float*)d_in[2];   // [256,1024]
    const float* W_bias    = (const float*)d_in[3];   // [1024]
    const float* kf        = (const float*)d_in[4];   // [4,256,1024]
    const float* rf        = (const float*)d_in[5];
    const float* bfw       = (const float*)d_in[6];   // [4,1024]
    const float* kb        = (const float*)d_in[7];
    const float* rb        = (const float*)d_in[8];
    const float* bbw       = (const float*)d_in[9];
    const float* ifh       = (const float*)d_in[10];  // [4,1,256]
    const float* ifc       = (const float*)d_in[11];
    const float* ibh       = (const float*)d_in[12];
    const float* ibc       = (const float*)d_in[13];
    const float* fck       = (const float*)d_in[14];  // [4,512,256]

    float* out_h = (float*)d_out;                          // [6,4096,256]
    float* out_c = out_h + (size_t)L_LEVELS * N_NODES * DIM;

    // -------- workspace bump allocator (256B aligned) --------
    char* base = (char*)d_ws;
    size_t off = 0;
    auto alloc = [&](size_t bytes) -> void* {
        void* p = base + off;
        off = (off + bytes + 255) & ~(size_t)255;
        return p;
    };
    const size_t KD  = (size_t)DIM * GATED;          // 256*1024
    const size_t SEQ = (size_t)N_NODES * MAX_CH * DIM;

    _Float16* wkT    = (_Float16*)alloc(KD * 2);                 // [1024][256]
    _Float16* krT[2][4];                                         // [1024][512]
    for (int d = 0; d < 2; ++d)
        for (int g = 0; g < 4; ++g) krT[d][g] = (_Float16*)alloc(2 * KD * 2);
    _Float16* fcT[4];                                            // [256][512]
    for (int g = 0; g < 4; ++g) fcT[g] = (_Float16*)alloc((size_t)2 * DIM * DIM * 2);
    _Float16* x16    = (_Float16*)alloc((size_t)L_LEVELS * N_NODES * DIM * 2);
    float*    htab   = (float*)alloc((size_t)(N_NODES + 1) * DIM * 4);
    float*    ctab   = (float*)alloc((size_t)(N_NODES + 1) * DIM * 4);
    _Float16* hs16   = (_Float16*)alloc(SEQ * 2);
    _Float16* hrev16 = (_Float16*)alloc(SEQ * 2);
    float*    cs     = (float*)alloc(SEQ * 4);
    int*      length = (int*)alloc((size_t)N_NODES * 4);
    float*    Wx     = (float*)alloc((size_t)N_NODES * GATED * 4);
    float*    Z      = (float*)alloc((size_t)N_NODES * GATED * 4);
    float*    hstf   = (float*)alloc((size_t)N_NODES * DIM * 4);
    float*    cstf   = (float*)alloc((size_t)N_NODES * DIM * 4);
    _Float16* hst16  = (_Float16*)alloc((size_t)N_NODES * DIM * 2);
    _Float16* y16    = (_Float16*)alloc(SEQ * 2 * 2);            // [N*8,512]
    _Float16* ylast  = (_Float16*)alloc((size_t)N_NODES * 2 * DIM * 2);
    float*    fcout0 = (float*)alloc(SEQ * 4);                   // [N*8,256]
    float*    fcout[4];
    fcout[0] = fcout0;
    for (int g = 1; g < 4; ++g) fcout[g] = (float*)alloc((size_t)N_NODES * DIM * 4);

    auto cvt = [&](const float* s, _Float16* d, size_t n) {
        cvt_f32_f16<<<(unsigned)((n + 255) / 256), 256, 0, stream>>>(s, d, (int)n);
    };
    auto cvtT = [&](const float* s, _Float16* d, int Ks, int Ns, int ldd, int kofs) {
        size_t n = (size_t)Ks * Ns;
        cvt_transpose_f16<<<(unsigned)((n + 255) / 256), 256, 0, stream>>>(
            s, d, Ks, Ns, ldd, kofs);
    };
    // GEMM launcher: C = [A1|A2] @ BT^T (+bias); BT is N-major
    auto gemm = [&](const _Float16* A1, int lda1, int K1,
                    const _Float16* A2, int lda2,
                    const _Float16* BT, int ldbt, const float* bias,
                    float* C, int ldc, int M, int N, int K) {
        if (!A2) { A2 = A1; lda2 = lda1; }       // never dereferenced when K1==K
        dim3 grid(N / 64, M / 64);
        gemm_f16_wmma<<<grid, 128, 0, stream>>>(A1, lda1, K1, A2, lda2,
                                                BT, ldbt, bias, C, ldc, M, N, K);
    };

    // -------- one-time staging: weights transposed to f16 N-major --------
    cvtT(W_kernel, wkT, DIM, GATED, DIM, 0);                 // [1024][256]
    for (int g = 0; g < 4; ++g) {
        cvtT(kf + g * KD, krT[0][g], DIM, GATED, 2 * DIM, 0);      // K rows
        cvtT(rf + g * KD, krT[0][g], DIM, GATED, 2 * DIM, DIM);    // R rows
        cvtT(kb + g * KD, krT[1][g], DIM, GATED, 2 * DIM, 0);
        cvtT(rb + g * KD, krT[1][g], DIM, GATED, 2 * DIM, DIM);
        cvtT(fck + (size_t)g * 2 * DIM * DIM, fcT[g], 2 * DIM, DIM, 2 * DIM, 0);
    }
    cvt(tensor, x16, (size_t)L_LEVELS * N_NODES * DIM);
    zero_row0<<<1, 256, 0, stream>>>(htab, ctab);

    // -------- tree levels (sequential by construction) --------
    for (int lvl = 0; lvl < L_LEVELS; ++lvl) {
        build_seq<<<N_NODES, 256, 0, stream>>>(
            indices + (size_t)lvl * N_NODES * MAX_CH,
            htab, ctab, hs16, hrev16, cs, length);

        // Wx = x_lvl @ W_kernel + W_bias          [4096,1024]
        gemm(x16 + (size_t)lvl * N_NODES * DIM, DIM, DIM,
             nullptr, 0, wkT, DIM, W_bias, Wx, GATED,
             N_NODES, GATED, DIM);

        for (int g = 0; g < 4; ++g) {
            for (int dir = 0; dir < 2; ++dir) {
                const float* ih = (dir ? ibh : ifh) + (size_t)g * DIM;
                const float* ic = (dir ? ibc : ifc) + (size_t)g * DIM;
                const float* bias = (dir ? bbw : bfw) + (size_t)g * GATED;
                const _Float16* seq = dir ? hrev16 : hs16;

                init_states<<<N_NODES, 256, 0, stream>>>(hstf, cstf, hst16, ih, ic);

                for (int t = 0; t < MAX_CH; ++t) {
                    // Z = [x_t ; h] @ [K ; R] + b    (fused K=512 GEMM)
                    gemm(seq + (size_t)t * DIM, MAX_CH * DIM, DIM,
                         hst16, DIM,
                         krT[dir][g], 2 * DIM, bias,
                         Z, GATED, N_NODES, GATED, 2 * DIM);
                    lstm_cell<<<N_NODES, 256, 0, stream>>>(
                        Z, hstf, cstf, hst16, y16, dir * DIM, t);
                }
            }
            if (g == 0) {
                // full-sequence FC:  [N*8,512] @ [512,256]
                gemm(y16, 2 * DIM, 2 * DIM, nullptr, 0,
                     fcT[g], 2 * DIM, nullptr, fcout0, DIM,
                     N_NODES * MAX_CH, DIM, 2 * DIM);
            } else {
                gather_last<<<dim3(N_NODES, 2), 256, 0, stream>>>(y16, length, ylast);
                gemm(ylast, 2 * DIM, 2 * DIM, nullptr, 0,
                     fcT[g], 2 * DIM, nullptr, fcout[g], DIM,
                     N_NODES, DIM, 2 * DIM);
            }
        }

        combine<<<N_NODES, 256, 0, stream>>>(
            Wx, fcout[0], fcout[1], fcout[2], fcout[3], cs, length,
            htab, ctab,
            out_h + (size_t)lvl * N_NODES * DIM,
            out_c + (size_t)lvl * N_NODES * DIM);
    }
}
